// FastformerAttention3d_5970004541507
// MI455X (gfx1250) — matrix operand order repaired and checked by reading, not compiled, not run on hardware
//
#include <hip/hip_runtime.h>

typedef __attribute__((ext_vector_type(16))) _Float16 v16h;
typedef __attribute__((ext_vector_type(8)))  _Float16 v8h;
typedef __attribute__((ext_vector_type(8)))  float    v8f;

#define BATCHN  8
#define SEQ     1024
#define DIM     512
#define HEADS   8
#define DHEAD   64
#define FSCALE  0.125f               // DIM_HEAD ** -0.5
#define ROWS    (BATCHN * SEQ)       // 8192
#define BH      (BATCHN * HEADS)     // 64

// ---------------------------------------------------------------------------
// CDNA5 async copy: global -> LDS, 16 bytes per lane, tracked by ASYNCcnt.
// LDS address = low 32 bits of the flat shared pointer (LDS aperture maps
// addr[31:0] directly to the LDS byte offset, ISA 10.2).
// ---------------------------------------------------------------------------
__device__ __forceinline__ void ff_async_ld_b128(void* lds_dst, const void* gsrc) {
    asm volatile("global_load_async_to_lds_b128 %0, %1, off"
                 :: "v"((unsigned)(unsigned long long)lds_dst), "v"(gsrc)
                 : "memory");
}

__device__ __forceinline__ void ff_wait_async0() {
#if defined(__has_builtin) && __has_builtin(__builtin_amdgcn_s_wait_asynccnt)
    __builtin_amdgcn_s_wait_asynccnt(0);
#else
    asm volatile("s_wait_asynccnt 0x0" ::: "memory");
#endif
}

// ---------------------------------------------------------------------------
// Generic f32 -> f16 conversion (staging for WMMA operands)
// ---------------------------------------------------------------------------
__global__ void ff_cvt_f16(const float* __restrict__ in, _Float16* __restrict__ out, int n) {
    int i = blockIdx.x * blockDim.x + threadIdx.x;
    if (i < n) out[i] = (_Float16)in[i];
}

// ---------------------------------------------------------------------------
// C[M][ldc] = A[M][K](f16) @ W[Ncols][K](f16)^T + bias[col] (+ resid[row][col])
// Block = 128 threads = 4 waves; each wave owns a 16(M) x 64(N) strip.
// The 64x32h B chunk shared by all 4 waves is staged into LDS with
// GLOBAL_LOAD_ASYNC_TO_LDS_B128 (double-buffered: DMA of chunk k+1 overlaps
// the WMMAs of chunk k; s_wait_asynccnt 0 + barrier publishes each chunk).
// A fragment (16x32 f16): lane l<16 holds row l, K = k0..k0+15,
// lane l+16 K = k0+16..k0+31. B fragment (32x16): lane n<16
// holds col n, K=k0..k0+15; lane n+16 K=k0+16..31. C: component i -> row
// i + 8*(lane>>4), col = lane&15.
// ---------------------------------------------------------------------------
__global__ void ff_gemm_f16_wmma(const _Float16* __restrict__ A,
                                 const _Float16* __restrict__ W,
                                 const float*    __restrict__ bias,
                                 const float*    __restrict__ resid, int ldr,
                                 float*          __restrict__ C, int ldc,
                                 int K) {
    __shared__ __align__(64) _Float16 Bsm[2][64 * 32];   // 8 KB double buffer

    const int tid   = threadIdx.x;
    const int lane  = tid & 31;
    const int wave  = tid >> 5;
    const int lh    = lane & 15;          // low-half lane id
    const int hh    = lane >> 4;          // 0 = lanes 0-15, 1 = lanes 16-31
    const int tileN = blockIdx.x * 64;
    const int m0    = blockIdx.y * 64 + wave * 16;

    // Each thread DMAs two 16B chunks of the 64x32h B tile per K-step.
    // chunk c in [0,256): row r = c>>2 (weight row tileN+r), half-offset (c&3)*8.
    const int c0 = tid * 2;
    const int r0 = c0 >> 2,      ho0 = (c0 & 3) * 8;
    const int r1 = (c0 + 1) >> 2, ho1 = ((c0 + 1) & 3) * 8;

    v8f acc[4];
    acc[0] = (v8f){}; acc[1] = (v8f){}; acc[2] = (v8f){}; acc[3] = (v8f){};

    const _Float16* arow = A + (size_t)(m0 + lh) * K;

    // prefetch K-chunk 0 into buffer 0
    ff_async_ld_b128(&Bsm[0][r0 * 32 + ho0], W + (size_t)(tileN + r0) * K + ho0);
    ff_async_ld_b128(&Bsm[0][r1 * 32 + ho1], W + (size_t)(tileN + r1) * K + ho1);

    int it = 0;
    for (int k0 = 0; k0 < K; k0 += 32, ++it) {
        ff_wait_async0();       // own DMA done (data resident in LDS)
        __syncthreads();        // publish to all 4 waves

        const int cur = it & 1;
        if (k0 + 32 < K) {      // overlap next chunk's DMA with this chunk's WMMAs
            const int nxt = cur ^ 1;
            const int kn  = k0 + 32;
            ff_async_ld_b128(&Bsm[nxt][r0 * 32 + ho0], W + (size_t)(tileN + r0) * K + kn + ho0);
            ff_async_ld_b128(&Bsm[nxt][r1 * 32 + ho1], W + (size_t)(tileN + r1) * K + kn + ho1);
        }

        union { v16h v; v8h h[2]; } a;
        const int kb = k0 + hh * 16;
        a.h[0] = *(const v8h*)(arow + kb);
        a.h[1] = *(const v8h*)(arow + kb + 8);

        const _Float16* bb = &Bsm[cur][0];
        v16h b[4];
#pragma unroll
        for (int j = 0; j < 4; ++j)
            b[j] = *(const v16h*)(bb + (16 * j + lh) * 32 + hh * 16);

#pragma unroll
        for (int j = 0; j < 4; ++j)
            acc[j] = __builtin_amdgcn_wmma_f32_16x16x32_f16(
                false, a.v, false, b[j], (short)0, acc[j], false, false);
    }

#pragma unroll
    for (int i = 0; i < 8; ++i) {
        const int row = m0 + i + 8 * hh;
#pragma unroll
        for (int j = 0; j < 4; ++j) {
            const int col = tileN + 16 * j + lh;
            float v = acc[j][i] + bias[col];
            if (resid) v += resid[(size_t)row * ldr + col];
            C[(size_t)row * ldc + col] = v;
        }
    }
}

// ---------------------------------------------------------------------------
// prep1: per (b,h,n) row -> f16 copy of q, and rowsums qs = sum_d q, ks = sum_d k
// tid = (b*H + h)*SEQ + n
// ---------------------------------------------------------------------------
__global__ void ff_prep1(const float* __restrict__ qkv,
                         _Float16* __restrict__ qh,
                         float* __restrict__ qs, float* __restrict__ ks) {
    int tid = blockIdx.x * blockDim.x + threadIdx.x;
    if (tid >= BH * SEQ) return;
    int b = tid >> 13;            // / (HEADS*SEQ)
    int h = (tid >> 10) & 7;
    int n = tid & (SEQ - 1);
    const float* qrow = qkv + ((size_t)(b * SEQ + n)) * (3 * DIM) + h * DHEAD;
    const float* krow = qrow + DIM;
    float sq = 0.f, sk = 0.f;
#pragma unroll 8
    for (int d = 0; d < DHEAD; ++d) {
        float qv = qrow[d];
        qh[(size_t)tid * DHEAD + d] = (_Float16)qv;
        sq += qv;
        sk += krow[d];
    }
    qs[tid] = sq;
    ks[tid] = sk;
}

// ---------------------------------------------------------------------------
// Streaming-softmax stage (used for stage 1 and stage 2):
//   outg[bh,n] = sum_m softmax_m((Ah[bh,n,:]·Wh[m,:] + bias[m])*SCALE) * sums[bh,m]
// One wave per (bh, 16-row tile). A fragments loaded once, reused over all 64
// m-tiles (2 WMMA each, K=64). Per-lane online softmax with value-dim = 1,
// then a 16-lane log-sum-exp merge (masks 1,2,4,8 stay within each half-wave,
// which holds rows 0-7 / 8-15 respectively).
// ---------------------------------------------------------------------------
__global__ void ff_stage_softmax_wmma(const _Float16* __restrict__ Ah,   // [BH][SEQ][64]
                                      const _Float16* __restrict__ Wh,   // [SEQ][64]
                                      const float*    __restrict__ bias, // [SEQ]
                                      const float*    __restrict__ sums, // [BH][SEQ]
                                      float*          __restrict__ outg) // [BH][SEQ]
{
    const int lane = threadIdx.x & 31;
    const int lh = lane & 15;
    const int hh = lane >> 4;
    const int n0 = blockIdx.x * 16;
    const int bh = blockIdx.y;

    const _Float16* arow = Ah + ((size_t)bh * SEQ + n0 + lh) * DHEAD;
    union { v16h v; v8h h[2]; } a0, a1;
    a0.h[0] = *(const v8h*)(arow + hh * 16);
    a0.h[1] = *(const v8h*)(arow + hh * 16 + 8);
    a1.h[0] = *(const v8h*)(arow + 32 + hh * 16);
    a1.h[1] = *(const v8h*)(arow + 32 + hh * 16 + 8);

    float mx[8], sm[8], aw[8];
#pragma unroll
    for (int i = 0; i < 8; ++i) { mx[i] = -1e30f; sm[i] = 0.f; aw[i] = 0.f; }

    for (int mt = 0; mt < SEQ; mt += 16) {
        const _Float16* wrow = Wh + (size_t)(mt + lh) * DHEAD + hh * 16;
        v16h b0 = *(const v16h*)(wrow);
        v16h b1 = *(const v16h*)(wrow + 32);
        v8f c = (v8f){};
        c = __builtin_amdgcn_wmma_f32_16x16x32_f16(false, a0.v, false, b0, (short)0, c, false, false);
        c = __builtin_amdgcn_wmma_f32_16x16x32_f16(false, a1.v, false, b1, (short)0, c, false, false);

        const float bcol = bias[mt + lh];
        const float sv   = sums[bh * SEQ + mt + lh];
#pragma unroll
        for (int i = 0; i < 8; ++i) {
            float l  = (c[i] + bcol) * FSCALE;
            float mn = fmaxf(mx[i], l);
            float e  = __expf(l - mn);
            float al = __expf(mx[i] - mn);
            sm[i] = sm[i] * al + e;
            aw[i] = aw[i] * al + e * sv;
            mx[i] = mn;
        }
    }

#pragma unroll
    for (int i = 0; i < 8; ++i) {
        float mi = mx[i], si = sm[i], ai = aw[i];
        for (int off = 1; off < 16; off <<= 1) {
            float om = __shfl_xor(mi, off, 32);
            float os = __shfl_xor(si, off, 32);
            float oa = __shfl_xor(ai, off, 32);
            float mn = fmaxf(mi, om);
            float e0 = __expf(mi - mn), e1 = __expf(om - mn);
            si = si * e0 + os * e1;
            ai = ai * e0 + oa * e1;
            mi = mn;
        }
        if (lh == 0)
            outg[(size_t)bh * SEQ + n0 + i + 8 * hh] = ai / si;   // sum A*val
    }
}

// ---------------------------------------------------------------------------
// prep2: p = q_global * k  (f16 for stage-2 A), ps = q_global * ks
// ---------------------------------------------------------------------------
__global__ void ff_prep2(const float* __restrict__ qkv,
                         const float* __restrict__ qg,
                         const float* __restrict__ ks,
                         _Float16* __restrict__ ph,
                         float* __restrict__ ps) {
    int tid = blockIdx.x * blockDim.x + threadIdx.x;
    if (tid >= BH * SEQ) return;
    int b = tid >> 13;
    int h = (tid >> 10) & 7;
    int n = tid & (SEQ - 1);
    const float* krow = qkv + ((size_t)(b * SEQ + n)) * (3 * DIM) + DIM + h * DHEAD;
    const float g = qg[tid];
#pragma unroll 8
    for (int d = 0; d < DHEAD; ++d)
        ph[(size_t)tid * DHEAD + d] = (_Float16)(g * krow[d]);
    ps[tid] = g * ks[tid];
}

// ---------------------------------------------------------------------------
// u formation: u[b,n, h*64+d] = k_global[b,h,n] * v[b,h,n,d]   (f16 for GEMM)
// ---------------------------------------------------------------------------
__global__ void ff_form_u(const float* __restrict__ qkv,
                          const float* __restrict__ kg,
                          _Float16* __restrict__ uh) {
    int tid = blockIdx.x * blockDim.x + threadIdx.x;
    if (tid >= ROWS * DIM) return;
    int row = tid >> 9;            // / DIM
    int c   = tid & (DIM - 1);
    int h   = c >> 6;
    int b   = row >> 10;
    int n   = row & (SEQ - 1);
    int bhn = ((b * HEADS + h) << 10) | n;
    float v = qkv[(size_t)row * (3 * DIM) + 2 * DIM + c];
    uh[tid] = (_Float16)(kg[bhn] * v);
}

// ---------------------------------------------------------------------------
extern "C" void kernel_launch(void* const* d_in, const int* in_sizes, int n_in,
                              void* d_out, int out_size, void* d_ws, size_t ws_size,
                              hipStream_t stream) {
    const float* x     = (const float*)d_in[0];   // [8,1024,512]
    const float* w_qkv = (const float*)d_in[1];   // [1536,512]
    const float* b_qkv = (const float*)d_in[2];   // [1536]
    const float* wq    = (const float*)d_in[3];   // [1024,64]
    const float* bq    = (const float*)d_in[4];   // [1024]
    const float* wk    = (const float*)d_in[5];   // [1024,64]
    const float* bk    = (const float*)d_in[6];   // [1024]
    const float* w_out = (const float*)d_in[7];   // [512,512]
    const float* b_out = (const float*)d_in[8];   // [512]
    float* out = (float*)d_out;                   // [8,1024,512]

    char* ws = (char*)d_ws;
    size_t ofs = 0;
    auto carve = [&](size_t bytes) -> void* {
        void* p = ws + ofs;
        ofs = (ofs + bytes + 255) & ~(size_t)255;
        return p;
    };

    float*    qkv    = (float*)   carve((size_t)ROWS * 3 * DIM * 4);   // 48 MB
    _Float16* xh     = (_Float16*)carve((size_t)ROWS * DIM * 2);       // 8 MB
    _Float16* wqkvh  = (_Float16*)carve((size_t)3 * DIM * DIM * 2);    // 1.5 MB
    _Float16* qh     = (_Float16*)carve((size_t)BH * SEQ * DHEAD * 2); // 8 MB
    _Float16* ph     = (_Float16*)carve((size_t)BH * SEQ * DHEAD * 2); // 8 MB
    _Float16* uh     = (_Float16*)carve((size_t)ROWS * DIM * 2);       // 8 MB
    _Float16* woh    = (_Float16*)carve((size_t)DIM * DIM * 2);        // 0.5 MB
    _Float16* wqh    = (_Float16*)carve((size_t)SEQ * DHEAD * 2);
    _Float16* wkh    = (_Float16*)carve((size_t)SEQ * DHEAD * 2);
    float*    qs     = (float*)   carve((size_t)BH * SEQ * 4);
    float*    ks     = (float*)   carve((size_t)BH * SEQ * 4);
    float*    qg     = (float*)   carve((size_t)BH * SEQ * 4);
    float*    ps     = (float*)   carve((size_t)BH * SEQ * 4);
    float*    kg     = (float*)   carve((size_t)BH * SEQ * 4);

    auto cvt = [&](const float* src, _Float16* dst, int n) {
        ff_cvt_f16<<<dim3((n + 255) / 256), dim3(256), 0, stream>>>(src, dst, n);
    };
    cvt(x,     xh,    ROWS * DIM);
    cvt(w_qkv, wqkvh, 3 * DIM * DIM);
    cvt(wq,    wqh,   SEQ * DHEAD);
    cvt(wk,    wkh,   SEQ * DHEAD);
    cvt(w_out, woh,   DIM * DIM);

    // qkv = x @ w_qkv^T + b_qkv  (M=8192, N=1536, K=512)
    ff_gemm_f16_wmma<<<dim3((3 * DIM) / 64, ROWS / 64), dim3(128), 0, stream>>>(
        xh, wqkvh, b_qkv, (const float*)nullptr, 0, qkv, 3 * DIM, DIM);

    // q -> f16, row sums of q and k
    ff_prep1<<<dim3((BH * SEQ) / 256), dim3(256), 0, stream>>>(qkv, qh, qs, ks);

    // stage 1: q_global[bh,n] = sum_m softmax((q·wq + bq)*s)[m] * qs[m]
    ff_stage_softmax_wmma<<<dim3(SEQ / 16, BH), dim3(32), 0, stream>>>(
        qh, wqh, bq, qs, qg);

    // p = q_global * k (f16), ps = q_global * ks
    ff_prep2<<<dim3((BH * SEQ) / 256), dim3(256), 0, stream>>>(qkv, qg, ks, ph, ps);

    // stage 2: k_global[bh,n] = sum_m softmax((p·wk + bk)*s)[m] * ps[m]
    ff_stage_softmax_wmma<<<dim3(SEQ / 16, BH), dim3(32), 0, stream>>>(
        ph, wkh, bk, ps, kg);

    // u = k_global * v  (f16)
    ff_form_u<<<dim3((ROWS * DIM) / 256), dim3(256), 0, stream>>>(qkv, kg, uh);

    // out = u @ w_out^T + b_out + qo   (M=8192, N=512, K=512; residual = q cols of qkv)
    ff_gemm_f16_wmma<<<dim3(DIM / 64, ROWS / 64), dim3(128), 0, stream>>>(
        uh, woh, b_out, /*resid=*/qkv, /*ldr=*/3 * DIM, out, DIM, DIM);
}
